// SingleHeadAttention_78752520339512
// MI455X (gfx1250) — matrix operand order repaired and checked by reading, not compiled, not run on hardware
//
#include <hip/hip_runtime.h>
#include <cmath>

typedef __attribute__((ext_vector_type(16))) _Float16 v16h;
typedef __attribute__((ext_vector_type(8)))  _Float16 v8h;
typedef __attribute__((ext_vector_type(8)))  float    v8f;
typedef __attribute__((ext_vector_type(4)))  int      v4i;

union V16U { v16h v; v8h h[2]; };

#define BATCH    4
#define SEQ      4096
#define EMBED    1024
#define HDIM     64
#define NQKV     192      // 3*HDIM
#define BR       64       // query rows per block (4 waves x 16)
#define BC       32       // key tile

// ---------------------------------------------------------------------------
// CDNA5 async global->LDS copy (ASYNCcnt path), with safe fallback.
// Builtin params are v4i pointers: AS1 (printed '__device__') src, AS3 dst.
// ---------------------------------------------------------------------------
#if __has_builtin(__builtin_amdgcn_global_load_async_to_lds_b128)
#define HAS_ASYNC 1
__device__ __forceinline__ void async_copy16(const _Float16* g, _Float16* l) {
    __builtin_amdgcn_global_load_async_to_lds_b128(
        (__attribute__((address_space(1))) v4i*)(g),
        (__attribute__((address_space(3))) v4i*)(l), 0, 0);
}
__device__ __forceinline__ void wait_async() {
#if __has_builtin(__builtin_amdgcn_s_wait_asynccnt)
    __builtin_amdgcn_s_wait_asynccnt(0);
#else
    asm volatile("s_wait_asynccnt 0x0" ::: "memory");
#endif
}
#else
#define HAS_ASYNC 0
__device__ __forceinline__ void async_copy16(const _Float16* g, _Float16* l) {
    *(v8h*)l = *(const v8h*)g;   // synchronous fallback
}
__device__ __forceinline__ void wait_async() {}
#endif

// ---------------------------------------------------------------------------
// Kernel 0: QKV weight [1024][192] f32 -> wT [192][1024] f16
// ---------------------------------------------------------------------------
__global__ void wT_cvt_kernel(const float* __restrict__ w, _Float16* __restrict__ wT) {
    int idx = blockIdx.x * 256 + threadIdx.x;        // 192*1024 total
    if (idx >= NQKV * EMBED) return;
    int n = idx % NQKV;
    int k = idx / NQKV;                               // consecutive threads: coalesced read
    wT[(size_t)n * EMBED + k] = (_Float16)w[(size_t)k * NQKV + n];
}

// ---------------------------------------------------------------------------
// Kernel 1: qkv = x @ W.  M=16384 K=1024 N=192.  Block = 64 rows, 4 waves.
// q (scaled 1/sqrt(64)) and k row-major [B*S][64] f16; v written TRANSPOSED
// as vT[b][d][s] f16 with packed 16B stores (C layout gives 8 consecutive
// rows per lane -> one v8h store per d-column).
// ---------------------------------------------------------------------------
__global__ void __launch_bounds__(128)
qkv_proj_kernel(const float* __restrict__ x, const _Float16* __restrict__ wT,
                _Float16* __restrict__ qb, _Float16* __restrict__ kb_,
                _Float16* __restrict__ vT) {
    __shared__ __align__(16) _Float16 xs[64][32 + 8];      // x chunk, f16
    __shared__ __align__(16) _Float16 ws[NQKV][32 + 8];    // W^T chunk: ws[col][k]

    const int t     = threadIdx.x;
    const int wv    = t >> 5;
    const int ln    = t & 31;
    const int lhalf = ln >> 4;
    const int lcol  = ln & 15;
    const size_t row0 = (size_t)blockIdx.x * 64;

    v8f zero = {};
    v8f acc[12];
#pragma unroll
    for (int n = 0; n < 12; ++n) acc[n] = zero;

    const int r  = t & 63;            // staging row for x
    const int cb = (t >> 6) * 16;     // staging col base for x
    const int kbA = lhalf * 16;
    const int kbB = lhalf * 16;

    for (int k0 = 0; k0 < EMBED; k0 += 32) {
        // stage x chunk (64 rows x 32 K) as f16
        {
            const float* src = x + (row0 + r) * EMBED + k0 + cb;
#pragma unroll
            for (int i = 0; i < 16; ++i) xs[r][cb + i] = (_Float16)src[i];
        }
        // stage W^T chunk (192 cols x 32 K)
        for (int col = t; col < NQKV; col += 128) {
            const _Float16* src = wT + (size_t)col * EMBED + k0;
            v8h a0 = *(const v8h*)(src);
            v8h a1 = *(const v8h*)(src + 8);
            v8h a2 = *(const v8h*)(src + 16);
            v8h a3 = *(const v8h*)(src + 24);
            *(v8h*)&ws[col][0]  = a0;
            *(v8h*)&ws[col][8]  = a1;
            *(v8h*)&ws[col][16] = a2;
            *(v8h*)&ws[col][24] = a3;
        }
        __syncthreads();

        // A fragment: 16x32, row = lane&15
        const int mrow = wv * 16 + lcol;
        V16U a;
        a.h[0] = *(const v8h*)&xs[mrow][kbA];
        a.h[1] = *(const v8h*)&xs[mrow][kbA + 8];

        // software-pipelined B fragments: load n+1 while WMMA n runs
        V16U bcur;
        {
            const _Float16* bp = &ws[lcol][kbB];
            bcur.h[0] = *(const v8h*)(bp);
            bcur.h[1] = *(const v8h*)(bp + 8);
        }
#pragma unroll
        for (int n = 0; n < 12; ++n) {
            V16U bnext = bcur;
            if (n < 11) {
                const _Float16* bp = &ws[(n + 1) * 16 + lcol][kbB];
                bnext.h[0] = *(const v8h*)(bp);
                bnext.h[1] = *(const v8h*)(bp + 8);
            }
            acc[n] = __builtin_amdgcn_wmma_f32_16x16x32_f16(
                false, a.v, false, bcur.v, (short)0, acc[n], false, false);
            bcur = bnext;
        }
        __syncthreads();
    }

    // write-out: C layout -> lane holds col (lane&15), rows mbase..mbase+7
    const int   mbase = lhalf * 8;
    const float scale = 0.125f;  // 1/sqrt(64), folded into q
    const size_t bb = row0 >> 12;           // batch index (blocks never straddle)
    const size_t s0 = (row0 & 4095) + wv * 16 + mbase;
#pragma unroll
    for (int n = 0; n < 12; ++n) {
        const int col = n * 16 + lcol;
        if (col < 128) {
#pragma unroll
            for (int j = 0; j < 8; ++j) {
                size_t row = row0 + wv * 16 + mbase + j;
                float  val = acc[n][j];
                if (col < 64) qb [row * HDIM + col]        = (_Float16)(val * scale);
                else          kb_[row * HDIM + (col - 64)] = (_Float16)val;
            }
        } else {
            const int dcol = col - 128;
            v8h pack;
#pragma unroll
            for (int j = 0; j < 8; ++j) pack[j] = (_Float16)acc[n][j];
            *(v8h*)&vT[(bb * HDIM + dcol) * SEQ + s0] = pack;
        }
    }
}

// ---------------------------------------------------------------------------
// Kernel 2: flash attention.  Block = 64 query rows (4 waves x 16), f16 WMMA,
// f32 accum, online softmax.  Double-buffered async K/V tiles of 32 keys.
// ---------------------------------------------------------------------------
__global__ void __launch_bounds__(128)
flash_attn_kernel(const _Float16* __restrict__ qg, const _Float16* __restrict__ kg,
                  const _Float16* __restrict__ vT, float* __restrict__ out) {
    __shared__ __align__(16) _Float16 Kbuf[2][BC][HDIM + 8];   // keys row-major
    __shared__ __align__(16) _Float16 Vbuf[2][HDIM][BC + 8];   // values transposed
    __shared__ __align__(16) _Float16 Pw[4][16][BC + 8];       // per-wave P buffer

    const int t     = threadIdx.x;
    const int wv    = t >> 5;
    const int ln    = t & 31;
    const int lhalf = ln >> 4;
    const int lcol  = ln & 15;
    const int mbase = lhalf * 8;
    const int b     = blockIdx.y;
    const int q0    = blockIdx.x * BR;
    const size_t base = (size_t)b * SEQ * HDIM;

    // Q fragments: rows wv*16 + lcol, two K=32 chunks of d
    const int kbA = lhalf * 16;
    const int kbB = lhalf * 16;
    const _Float16* qp = qg + base + (size_t)(q0 + wv * 16 + lcol) * HDIM;
    v16h qa[2];
#pragma unroll
    for (int c = 0; c < 2; ++c) {
        V16U u;
        u.h[0] = *(const v8h*)(qp + c * 32 + kbA);
        u.h[1] = *(const v8h*)(qp + c * 32 + kbA + 8);
        qa[c] = u.v;
    }

    float m[8], l[8];
    v8f zero = {};
    v8f acc[4];
#pragma unroll
    for (int j = 0; j < 8; ++j) { m[j] = -INFINITY; l[j] = 0.f; }
#pragma unroll
    for (int n = 0; n < 4; ++n) acc[n] = zero;

    // staging roles
    const int kK = t & 31;            // K: key row
    const int dK = (t >> 5) * 16;     // K: d segment (0/16/32/48)
    const int dV = t & 63;            // V: d row
    const int sV = (t >> 6) * 16;     // V: key segment (0/16)
    const _Float16* kbase_g = kg + base;
    const _Float16* vbase_g = vT + (size_t)b * HDIM * SEQ;

#define ISSUE_TILE(BUF, KT)                                                    \
    do {                                                                       \
        const _Float16* ksrc = kbase_g + (size_t)((KT) + kK) * HDIM + dK;      \
        async_copy16(ksrc,     &Kbuf[BUF][kK][dK]);                            \
        async_copy16(ksrc + 8, &Kbuf[BUF][kK][dK + 8]);                        \
        const _Float16* vsrc = vbase_g + (size_t)dV * SEQ + (KT) + sV;         \
        async_copy16(vsrc,     &Vbuf[BUF][dV][sV]);                            \
        async_copy16(vsrc + 8, &Vbuf[BUF][dV][sV + 8]);                        \
    } while (0)

    ISSUE_TILE(0, 0);

    for (int it = 0; it < SEQ / BC; ++it) {
        const int p = it & 1;
        wait_async();
        __syncthreads();                       // tile p landed; all waves done with buf p^1
        if (it + 1 < SEQ / BC) ISSUE_TILE(p ^ 1, (it + 1) * BC);

        // ---- S = Q K^T : two 16x16 subtiles, each 2 WMMAs over d ----
        v8f s[2];
#pragma unroll
        for (int sub = 0; sub < 2; ++sub) {
            v8f sv = zero;
#pragma unroll
            for (int c = 0; c < 2; ++c) {
                V16U u;
                const _Float16* bp = &Kbuf[p][sub * 16 + lcol][c * 32 + kbB];
                u.h[0] = *(const v8h*)(bp);
                u.h[1] = *(const v8h*)(bp + 8);
                sv = __builtin_amdgcn_wmma_f32_16x16x32_f16(
                    false, qa[c], false, u.v, (short)0, sv, false, false);
            }
            s[sub] = sv;
        }

        // ---- online softmax (row stats live in 16-lane groups) ----
        float p0[8], p1[8];
#pragma unroll
        for (int j = 0; j < 8; ++j) {
            float rm = fmaxf(s[0][j], s[1][j]);
            rm = fmaxf(rm, __shfl_xor(rm, 1, 32));
            rm = fmaxf(rm, __shfl_xor(rm, 2, 32));
            rm = fmaxf(rm, __shfl_xor(rm, 4, 32));
            rm = fmaxf(rm, __shfl_xor(rm, 8, 32));
            float mn   = fmaxf(m[j], rm);
            float corr = __expf(m[j] - mn);
            m[j] = mn;
            float e0 = __expf(s[0][j] - mn);
            float e1 = __expf(s[1][j] - mn);
            float rs = e0 + e1;
            rs += __shfl_xor(rs, 1, 32);
            rs += __shfl_xor(rs, 2, 32);
            rs += __shfl_xor(rs, 4, 32);
            rs += __shfl_xor(rs, 8, 32);
            l[j] = l[j] * corr + rs;
#pragma unroll
            for (int n = 0; n < 4; ++n) acc[n][j] *= corr;
            p0[j] = e0;
            p1[j] = e1;
        }

        // ---- P: C-layout -> LDS -> A-layout (wave-local, LDS in-order) ----
#pragma unroll
        for (int j = 0; j < 8; ++j) {
            Pw[wv][mbase + j][lcol]      = (_Float16)p0[j];
            Pw[wv][mbase + j][16 + lcol] = (_Float16)p1[j];
        }
        V16U pa;
        {
            const _Float16* pp = &Pw[wv][lcol][0];
            pa.h[0] = *(const v8h*)(pp + kbA);
            pa.h[1] = *(const v8h*)(pp + kbA + 8);
        }

        // ---- O += P @ V : 4 d-tiles, K=32 ----
#pragma unroll
        for (int n = 0; n < 4; ++n) {
            V16U u;
            const _Float16* bp = &Vbuf[p][n * 16 + lcol][kbB];
            u.h[0] = *(const v8h*)(bp);
            u.h[1] = *(const v8h*)(bp + 8);
            acc[n] = __builtin_amdgcn_wmma_f32_16x16x32_f16(
                false, pa.v, false, u.v, (short)0, acc[n], false, false);
        }
    }
#undef ISSUE_TILE

    // ---- epilogue: out = acc / l ----
    float inv[8];
#pragma unroll
    for (int j = 0; j < 8; ++j) inv[j] = 1.f / l[j];
    float* op = out + base + (size_t)(q0 + wv * 16 + mbase) * HDIM;
#pragma unroll
    for (int n = 0; n < 4; ++n)
#pragma unroll
        for (int j = 0; j < 8; ++j)
            op[(size_t)j * HDIM + n * 16 + lcol] = acc[n][j] * inv[j];
}

// ---------------------------------------------------------------------------
extern "C" void kernel_launch(void* const* d_in, const int* in_sizes, int n_in,
                              void* d_out, int out_size, void* d_ws, size_t ws_size,
                              hipStream_t stream) {
    (void)in_sizes; (void)n_in; (void)out_size; (void)ws_size;
    const float* x = (const float*)d_in[0];
    const float* w = (const float*)d_in[1];
    float* out = (float*)d_out;

    char* ws = (char*)d_ws;
    const size_t WT_BYTES  = (size_t)NQKV * EMBED * sizeof(_Float16);        // 384 KB
    const size_t QKV_BYTES = (size_t)BATCH * SEQ * HDIM * sizeof(_Float16);  // 2 MB each
    _Float16* wT = (_Float16*)(ws);
    _Float16* q  = (_Float16*)(ws + WT_BYTES);
    _Float16* k  = (_Float16*)(ws + WT_BYTES + QKV_BYTES);
    _Float16* vT = (_Float16*)(ws + WT_BYTES + 2 * QKV_BYTES);

    wT_cvt_kernel<<<(NQKV * EMBED + 255) / 256, 256, 0, stream>>>(w, wT);
    qkv_proj_kernel<<<(BATCH * SEQ) / 64, 128, 0, stream>>>(x, wT, q, k, vT);
    flash_attn_kernel<<<dim3(SEQ / BR, BATCH), 128, 0, stream>>>(q, k, vT, out);
}
